// CNN_65395172049280
// MI455X (gfx1250) — matrix operand order repaired and checked by reading, not compiled, not run on hardware
//
#include <hip/hip_runtime.h>

// Problem constants (from reference)
#define V_  50000
#define E_  100
#define L_  2500
#define B_  8
#define F_  50
#define Y_  8921
#define K_  9
#define PAD_ 4
#define FP  64                 // padded feature dim; col 50 = ones column (denominator)
#define LP_ 2560               // padded L for the f-major transposed copy
#define NYT ((Y_ + 15) / 16)   // 558 label tiles
#define NCHF (L_ / 32)         // 78 full L-chunks
#define NCH ((L_ + 31) / 32)   // 79 total chunks (1 tail)
#define LOG2E 1.4426950408889634f

typedef __attribute__((ext_vector_type(16))) _Float16 v16h;
typedef __attribute__((ext_vector_type(8)))  float    v8f;

// ---------------------------------------------------------------------------
// Kernel 1: embedding gather + Conv1d + tanh.
// Writes BOTH layouts (coalesced via an LDS transpose tile):
//   hpad[b][l][64]    f16 (l-major, scores-GEMM B operand; f>50 zero, f==50 = 1)
//   hT  [b][64][2560] f16 (f-major, m-GEMM B operand; row 50 = ones column)
// ---------------------------------------------------------------------------
__global__ __launch_bounds__(256) void conv_embed_kernel(
    const int* __restrict__ x, const float* __restrict__ W_embed,
    const float* __restrict__ conv_w, const float* __restrict__ conv_b,
    _Float16* __restrict__ hpad, _Float16* __restrict__ hT) {
  __shared__ _Float16 tileT[64][72];   // [f][l_local]; 144B row stride (16B aligned)
  const int b  = blockIdx.y;
  const int l0 = blockIdx.x * 64;

  for (int i = threadIdx.x; i < 64 * 64; i += 256) {
    int f  = i & 63;
    int ll = i >> 6;
    int l  = l0 + ll;
    float h = 0.f;
    if (l < L_) {
      if (f < F_) {
        float acc = conv_b[f];
        #pragma unroll
        for (int k = 0; k < K_; ++k) {
          int lp = l + k - PAD_;
          if (lp < 0 || lp >= L_) continue;
          int tok = x[b * L_ + lp];
          const float* er = W_embed + (long)tok * E_;
          const float* wr = conv_w + f * (E_ * K_) + k;   // conv_w[f][e][k]
          #pragma unroll 4
          for (int e = 0; e < E_; ++e) acc = fmaf(er[e], wr[e * K_], acc);
        }
        h = tanhf(acc);
      } else if (f == F_) {
        h = 1.0f;   // ones column -> softmax denominator falls out of the m-GEMM
      }
      hpad[((size_t)b * L_ + l) * FP + f] = (_Float16)h;
    }
    tileT[f][ll] = (_Float16)h;   // l >= L_ entries are 0
  }
  __syncthreads();

  for (int c = threadIdx.x; c < 64 * 8; c += 256) {   // transposed, coalesced b128s
    int f = c >> 3, lp = (c & 7) * 8;
    const float4 v = *(const float4*)&tileT[f][lp];
    *(float4*)&hT[((size_t)b * FP + f) * LP_ + l0 + lp] = v;
  }
}

// ---------------------------------------------------------------------------
// Kernel 2: fused label-wise attention, fixed-shift softmax in exp2 domain.
// block = (16-label tile, batch b); 256 threads = 8 waves striding L-chunks.
// ---------------------------------------------------------------------------
__global__ __launch_bounds__(256) void attn_kernel(
    const _Float16* __restrict__ hpad, const _Float16* __restrict__ hT,
    const float* __restrict__ U_w, const float* __restrict__ final_w,
    const float* __restrict__ final_b, float* __restrict__ out) {
  __shared__ _Float16 U16[16 * FP];     // U tile * log2(e), f16, zero-padded
  __shared__ _Float16 Pst[8][16 * 32];  // per-wave P staging (D-layout -> A-layout)
  __shared__ float    sM[16 * FP];      // merged numerator (col 50 = denominator)
  __shared__ float    mrow[16];         // fixed shift per label: ||U_y||_1 * log2(e)

  const int tid  = threadIdx.x;
  const int lane = tid & 31;
  const int wv   = tid >> 5;
  const int hi   = (lane >> 4) & 1;
  const int ln   = lane & 15;
  const int ytile = blockIdx.x;
  const int b     = blockIdx.y;

  for (int i = tid; i < 16 * FP; i += 256) {
    int r = i >> 6, c = i & (FP - 1);
    int y = ytile * 16 + r;
    float v = (y < Y_ && c < F_) ? U_w[y * F_ + c] * LOG2E : 0.f;
    U16[i] = (_Float16)v;
    sM[i] = 0.f;
  }
  if (tid < 16) {               // safe shift: S2 <= ||U2_y||_1 since |tanh|<=1
    int y = ytile * 16 + tid;
    float s = 0.f;
    if (y < Y_)
      for (int c = 0; c < F_; ++c) s += fabsf(U_w[y * F_ + c]) * LOG2E;
    mrow[tid] = s;
  }
  __syncthreads();

  // A operands from U tile (contiguous k order, same as B: row m=ln, K {c0..c0+15}, c0 = hi*16)
  v16h aU0, aU1;
  {
    const _Float16* row = U16 + ln * FP;
    const int c0 = hi * 16;
    #pragma unroll
    for (int j = 0; j < 8; ++j) {
      aU0[j]     = row[c0 + j];
      aU0[8 + j] = row[c0 + 8 + j];
      aU1[j]     = row[32 + c0 + j];
      aU1[8 + j] = row[32 + c0 + 8 + j];
    }
  }
  float nb2[8];                                  // -shift, ready for one add
  #pragma unroll
  for (int r = 0; r < 8; ++r) nb2[r] = -mrow[r + hi * 8];

  const _Float16* hb  = hpad + (size_t)b * L_ * FP;
  const _Float16* htb = hT   + (size_t)b * FP * LP_;
  _Float16* ps = Pst[wv];
  const int k0 = hi * 16;       // B k order (contiguous, same as A): lanes0-15 K=0..15, lanes16-31 K=16..31

  v8f Mac[4];
  #pragma unroll
  for (int t = 0; t < 4; ++t) Mac[t] = (v8f){};

  // ================= main loop: full chunks, no clamps, no masks =============
  for (int ch = wv; ch < NCHF; ch += 8) {
    const int l0 = ch * 32;
    const _Float16* ra = hb + (size_t)(l0 + ln) * FP;
    const _Float16* rb = ra + 16 * FP;
    __builtin_prefetch(ra + 8 * 32 * FP, 0, 3);           // next chunk of this wave
    __builtin_prefetch(rb + 8 * 32 * FP, 0, 3);

    v8f Sa = (v8f){}, Sb = (v8f){};
    {
      v16h Ba0, Ba1, Bb0, Bb1;
      #pragma unroll
      for (int j = 0; j < 16; ++j) {
        Ba0[j] = ra[k0 + j];  Ba1[j] = ra[32 + k0 + j];
        Bb0[j] = rb[k0 + j];  Bb1[j] = rb[32 + k0 + j];
      }
      Sa = __builtin_amdgcn_wmma_f32_16x16x32_f16(false, aU0, false, Ba0, (short)0, Sa, false, false);
      Sa = __builtin_amdgcn_wmma_f32_16x16x32_f16(false, aU1, false, Ba1, (short)0, Sa, false, false);
      Sb = __builtin_amdgcn_wmma_f32_16x16x32_f16(false, aU0, false, Bb0, (short)0, Sb, false, false);
      Sb = __builtin_amdgcn_wmma_f32_16x16x32_f16(false, aU1, false, Bb1, (short)0, Sb, false, false);
    }
    // P = exp2(S2 - m2): one add + native v_exp_f32 + cvt per element
    #pragma unroll
    for (int r = 0; r < 8; ++r) {
      int row = r + hi * 8;
      ps[row * 32 + ln]      = (_Float16)__builtin_amdgcn_exp2f(Sa[r] + nb2[r]);
      ps[row * 32 + 16 + ln] = (_Float16)__builtin_amdgcn_exp2f(Sb[r] + nb2[r]);
    }
    asm volatile("s_wait_dscnt 0" ::: "memory");          // LDS RAW within wave
    v16h aP;
    {
      const _Float16* row = ps + ln * 32;
      const int c0 = hi * 16;
      #pragma unroll
      for (int j = 0; j < 8; ++j) { aP[j] = row[c0 + j]; aP[8 + j] = row[c0 + 8 + j]; }
    }
    #pragma unroll
    for (int t = 0; t < 4; ++t) {
      const _Float16* col = htb + (size_t)(t * 16 + ln) * LP_ + l0 + k0;
      __builtin_prefetch(col + 8 * 32, 0, 3);
      v16h Bh;
      #pragma unroll
      for (int j = 0; j < 16; ++j) Bh[j] = col[j];
      Mac[t] = __builtin_amdgcn_wmma_f32_16x16x32_f16(false, aP, false, Bh, (short)0, Mac[t], false, false);
    }
  }

  // ================= tail chunk (l0 = 2496): clamp + mask ====================
  if (NCH > NCHF && wv == (NCHF & 7)) {
    const int l0 = NCHF * 32;
    v8f Sa = (v8f){}, Sb = (v8f){};
    {
      const _Float16* ra = hb + (size_t)min(l0 + ln, L_ - 1) * FP;
      const _Float16* rb = hb + (size_t)min(l0 + 16 + ln, L_ - 1) * FP;
      v16h Ba0, Ba1, Bb0, Bb1;
      #pragma unroll
      for (int j = 0; j < 16; ++j) {
        Ba0[j] = ra[k0 + j];  Ba1[j] = ra[32 + k0 + j];
        Bb0[j] = rb[k0 + j];  Bb1[j] = rb[32 + k0 + j];
      }
      Sa = __builtin_amdgcn_wmma_f32_16x16x32_f16(false, aU0, false, Ba0, (short)0, Sa, false, false);
      Sa = __builtin_amdgcn_wmma_f32_16x16x32_f16(false, aU1, false, Ba1, (short)0, Sa, false, false);
      Sb = __builtin_amdgcn_wmma_f32_16x16x32_f16(false, aU0, false, Bb0, (short)0, Sb, false, false);
      Sb = __builtin_amdgcn_wmma_f32_16x16x32_f16(false, aU1, false, Bb1, (short)0, Sb, false, false);
    }
    const bool va = (l0 + ln < L_), vb = (l0 + 16 + ln < L_);
    #pragma unroll
    for (int r = 0; r < 8; ++r) {
      int row = r + hi * 8;
      ps[row * 32 + ln]      = (_Float16)(va ? __builtin_amdgcn_exp2f(Sa[r] + nb2[r]) : 0.f);
      ps[row * 32 + 16 + ln] = (_Float16)(vb ? __builtin_amdgcn_exp2f(Sb[r] + nb2[r]) : 0.f);
    }
    asm volatile("s_wait_dscnt 0" ::: "memory");
    v16h aP;
    {
      const _Float16* row = ps + ln * 32;
      const int c0 = hi * 16;
      #pragma unroll
      for (int j = 0; j < 8; ++j) { aP[j] = row[c0 + j]; aP[8 + j] = row[c0 + 8 + j]; }
    }
    #pragma unroll
    for (int t = 0; t < 4; ++t) {
      const _Float16* col = htb + (size_t)(t * 16 + ln) * LP_ + l0 + k0;  // LP_ pads OOB
      v16h Bh;
      #pragma unroll
      for (int j = 0; j < 16; ++j) Bh[j] = col[j];
      Mac[t] = __builtin_amdgcn_wmma_f32_16x16x32_f16(false, aP, false, Bh, (short)0, Mac[t], false, false);
    }
  }

  // ---- cross-wave merge: plain sum (shift is a global constant per row)
  #pragma unroll
  for (int r = 0; r < 8; ++r) {
    int row = r + hi * 8;
    #pragma unroll
    for (int t = 0; t < 4; ++t)
      atomicAdd(&sM[row * FP + t * 16 + ln], Mac[t][r]);
  }
  __syncthreads();

  // ---- logits: final_w . (num/den) + final_b ; den = sM[row][50]
  {
    int row = tid >> 4, fl = tid & 15;
    int y = ytile * 16 + row;
    int yc = min(y, Y_ - 1);
    float acc = 0.f;
    for (int f = fl; f < F_; f += 16)
      acc += sM[row * FP + f] * final_w[yc * F_ + f];
    #pragma unroll
    for (int off = 8; off; off >>= 1) acc += __shfl_xor(acc, off, 32);
    if (fl == 0 && y < Y_)
      out[(size_t)b * Y_ + y] = acc / sM[row * FP + F_] + final_b[y];
  }
}

// ---------------------------------------------------------------------------
// Kernel 3: BCEWithLogits mean reduction over B*Y logits -> out[B*Y]
// ---------------------------------------------------------------------------
__global__ __launch_bounds__(512) void loss_kernel(const float* __restrict__ target,
                                                   float* __restrict__ out) {
  const int n = B_ * Y_;
  float acc = 0.f;
  for (int i = threadIdx.x; i < n; i += 512) {
    float yv = out[i], t = target[i];
    acc += fmaxf(yv, 0.f) - yv * t + log1pf(__expf(-fabsf(yv)));
  }
  __shared__ float red[16];
  #pragma unroll
  for (int off = 16; off; off >>= 1) acc += __shfl_xor(acc, off, 32);
  int lane = threadIdx.x & 31, w = threadIdx.x >> 5;
  if (lane == 0) red[w] = acc;
  __syncthreads();
  if (threadIdx.x == 0) {
    float s = 0.f;
    #pragma unroll
    for (int i = 0; i < 16; ++i) s += red[i];
    out[n] = s / (float)n;
  }
}

// ---------------------------------------------------------------------------
extern "C" void kernel_launch(void* const* d_in, const int* in_sizes, int n_in,
                              void* d_out, int out_size, void* d_ws, size_t ws_size,
                              hipStream_t stream) {
  const int*   x       = (const int*)d_in[0];
  const float* target  = (const float*)d_in[1];
  const float* W_embed = (const float*)d_in[2];
  const float* conv_w  = (const float*)d_in[3];
  const float* conv_b  = (const float*)d_in[4];
  const float* U_w     = (const float*)d_in[5];
  const float* final_w = (const float*)d_in[6];
  const float* final_b = (const float*)d_in[7];
  float* out = (float*)d_out;                       // [B*Y logits | loss]

  _Float16* hpad = (_Float16*)d_ws;                                      // 2.56 MB
  _Float16* hT   = (_Float16*)((char*)d_ws + (size_t)B_ * L_ * FP * 2);  // 2.62 MB

  conv_embed_kernel<<<dim3((L_ + 63) / 64, B_), 256, 0, stream>>>(
      x, W_embed, conv_w, conv_b, hpad, hT);
  attn_kernel<<<dim3(NYT, B_), 256, 0, stream>>>(hpad, hT, U_w, final_w, final_b, out);
  loss_kernel<<<1, 512, 0, stream>>>(target, out);
}
